// DeltaNetAttention_24507083391403
// MI455X (gfx1250) — compile-verified, matrix operand order checked by reading, not run on hardware
//
#include <hip/hip_runtime.h>

#define DEV static __device__ __forceinline__

typedef __attribute__((ext_vector_type(2)))  float          floatx2;
typedef __attribute__((ext_vector_type(8)))  float          floatx8;
typedef __attribute__((ext_vector_type(16))) __bf16         bf16x16;
typedef __attribute__((ext_vector_type(8)))  unsigned short ushort8;

constexpr int Bc  = 2;
constexpr int Tc  = 2048;
constexpr int Dc  = 1024;
constexpr int Hc  = 16;
constexpr int DHc = 64;
constexpr int MTOT = Bc * Tc;   // 4096 rows

// ---------------- helpers ----------------

DEV unsigned short f2bfu(float f) {
  unsigned u = __builtin_bit_cast(unsigned, f);
  unsigned r = u + 0x7FFFu + ((u >> 16) & 1u);   // round-to-nearest-even
  return (unsigned short)(r >> 16);
}

DEV unsigned int pack2bf(float lo, float hi) {
  return (unsigned int)f2bfu(lo) | ((unsigned int)f2bfu(hi) << 16);
}

DEV float silu(float u) { return u / (1.f + __expf(-u)); }

// bf16 WMMA wrapper: D = A(16x32) * B(32x16) + C, f32 accum
DEV floatx8 wmma_bf16(bf16x16 a, bf16x16 b, floatx8 c) {
  return __builtin_amdgcn_wmma_f32_16x16x32_bf16(false, a, false, b, (short)0, c, false, false);
}

// f32 WMMA wrapper: D = A(16x4) * B(4x16) + C
DEV floatx8 wmma_f32k4(floatx2 a, floatx2 b, floatx8 c) {
  return __builtin_amdgcn_wmma_f32_16x16x4_f32(false, a, false, b, (short)0, c, false, false);
}

// A-fragment (16x32 bf16) from row-major ushort LDS tile.
// lanes 0-15: M=lane, K = k0..k0+7 and k0+16..k0+23
// lanes 16-31: M=lane-16, K = k0+8..15 and k0+24..31
DEV bf16x16 fragBF(const unsigned short* tile, int ld, int r0, int k0, int lane) {
  int m  = lane & 15;
  int ko = (lane >> 4) * 8;
  const unsigned short* p = tile + (r0 + m) * ld + k0 + ko;
  union { bf16x16 v; ushort8 h[2]; } u;
  u.h[0] = *(const ushort8*)(p);
  u.h[1] = *(const ushort8*)(p + 16);
  return u.v;
}

// f32 16x16x4 fragments from f32 LDS tiles
// A rows: A[m][k], lanes: M=lane&15, K pair = k + 2*(lane>=16)
DEV floatx2 fragA_rows(const float* tile, int ld, int r0, int k, int lane) {
  int m = lane & 15; int ko = (lane >> 4) * 2;
  floatx2 a; a[0] = tile[(r0 + m) * ld + k + ko]; a[1] = tile[(r0 + m) * ld + k + ko + 1];
  return a;
}
// A = tile^T rows (i.e. column c0+m of tile)
DEV floatx2 fragA_cols(const float* tile, int ld, int c0, int k, int lane) {
  int m = lane & 15; int ko = (lane >> 4) * 2;
  floatx2 a; a[0] = tile[(k + ko) * ld + c0 + m]; a[1] = tile[(k + ko + 1) * ld + c0 + m];
  return a;
}
// B[kk][n] taken directly from tile (column n0+n)
DEV floatx2 fragB_cols(const float* tile, int ld, int k, int n0, int lane) {
  int n = lane & 15; int ko = (lane >> 4) * 2;
  floatx2 b; b[0] = tile[(k + ko) * ld + n0 + n]; b[1] = tile[(k + ko + 1) * ld + n0 + n];
  return b;
}
// B = tile^T : B[kk][n] = tile[r0+n][kk]  (used for K^T / Q K^T)
DEV floatx2 fragB_rows(const float* tile, int ld, int r0, int k, int lane) {
  int n = lane & 15; int ko = (lane >> 4) * 2;
  floatx2 b; b[0] = tile[(r0 + n) * ld + k + ko]; b[1] = tile[(r0 + n) * ld + k + ko + 1];
  return b;
}

// ---------------- kernel 1/6: WMMA GEMM, C = A[4096x1024] @ W[1024x1024] (x3 via blockIdx.z) ----------------

__global__ __launch_bounds__(256) void gemm_bf16_x3(
    const float* __restrict__ A,
    const float* __restrict__ W0, const float* __restrict__ W1, const float* __restrict__ W2,
    float* __restrict__ C0, float* __restrict__ C1, float* __restrict__ C2) {
  const float* W = (blockIdx.z == 0) ? W0 : (blockIdx.z == 1 ? W1 : W2);
  float*       C = (blockIdx.z == 0) ? C0 : (blockIdx.z == 1 ? C1 : C2);

  __shared__ __align__(16) unsigned short As[128][64];   // A tile, row-major (M x K)
  __shared__ __align__(16) unsigned short Bs[64][64];    // W tile transposed (N x K)

  const int tid  = threadIdx.x;
  const int lane = tid & 31, wave = tid >> 5;
  const int m0 = blockIdx.y * 128, n0 = blockIdx.x * 64;
  const int wm = (wave >> 1) * 32, wn = (wave & 1) * 32; // 4x2 wave grid, 32x32 per wave

  floatx8 acc[2][2] = {};

  for (int k0 = 0; k0 < Dc; k0 += 64) {
    { // stage A tile: 128x64, each thread 32 floats -> 4x b128 packed bf16 stores
      int r = tid >> 1, c = (tid & 1) * 32;
      const float* src = A + (size_t)(m0 + r) * Dc + k0 + c;
      uint4* dst = (uint4*)&As[r][c];
      #pragma unroll
      for (int i = 0; i < 4; i++) {
        float4 f0 = *(const float4*)(src + i * 8);
        float4 f1 = *(const float4*)(src + i * 8 + 4);
        uint4 p;
        p.x = pack2bf(f0.x, f0.y); p.y = pack2bf(f0.z, f0.w);
        p.z = pack2bf(f1.x, f1.y); p.w = pack2bf(f1.z, f1.w);
        dst[i] = p;
      }
    }
    { // stage W tile transposed: each thread covers K-pair x 8 cols -> packed b32 stores
      int kk2 = (tid & 31) * 2, nb = (tid >> 5) * 8;
      const float* src0 = W + (size_t)(k0 + kk2) * Dc + n0 + nb;
      const float* src1 = src0 + Dc;
      float4 a0 = *(const float4*)(src0);
      float4 a1 = *(const float4*)(src0 + 4);
      float4 b0 = *(const float4*)(src1);
      float4 b1 = *(const float4*)(src1 + 4);
      const float* ap = (const float*)&a0;
      const float* bp = (const float*)&b0;
      #pragma unroll
      for (int j = 0; j < 4; j++)
        *(unsigned int*)&Bs[nb + j][kk2] = pack2bf(ap[j], bp[j]);
      ap = (const float*)&a1; bp = (const float*)&b1;
      #pragma unroll
      for (int j = 0; j < 4; j++)
        *(unsigned int*)&Bs[nb + 4 + j][kk2] = pack2bf(ap[j], bp[j]);
    }
    __syncthreads();
    #pragma unroll
    for (int kk = 0; kk < 64; kk += 32) {
      bf16x16 a0 = fragBF(&As[0][0], 64, wm + 0,  kk, lane);
      bf16x16 a1 = fragBF(&As[0][0], 64, wm + 16, kk, lane);
      bf16x16 b0 = fragBF(&Bs[0][0], 64, wn + 0,  kk, lane);
      bf16x16 b1 = fragBF(&Bs[0][0], 64, wn + 16, kk, lane);
      acc[0][0] = wmma_bf16(a0, b0, acc[0][0]);
      acc[0][1] = wmma_bf16(a0, b1, acc[0][1]);
      acc[1][0] = wmma_bf16(a1, b0, acc[1][0]);
      acc[1][1] = wmma_bf16(a1, b1, acc[1][1]);
    }
    __syncthreads();
  }

  const int nL = lane & 15, mOff = (lane >> 4) * 8;
  #pragma unroll
  for (int i = 0; i < 2; i++)
    #pragma unroll
    for (int j = 0; j < 2; j++) {
      float* cp = C + (size_t)(m0 + wm + 16 * i + mOff) * Dc + n0 + wn + 16 * j + nL;
      #pragma unroll
      for (int r = 0; r < 8; r++) cp[(size_t)r * Dc] = acc[i][j][r];
    }
}

// ---------------- kernel 2/6: beta = sigmoid(x @ Wb) ----------------

__global__ __launch_bounds__(256) void beta_sigmoid_kernel(
    const float* __restrict__ x, const float* __restrict__ Wb, float* __restrict__ beta) {
  int idx = blockIdx.x * 256 + threadIdx.x;     // [m*16 + h]
  int h = idx & (Hc - 1), m = idx >> 4;
  const float* xr = x + (size_t)m * Dc;
  float acc = 0.f;
  #pragma unroll 4
  for (int kk = 0; kk < Dc; kk++) acc = fmaf(xr[kk], Wb[kk * Hc + h], acc);
  beta[idx] = 1.f / (1.f + __expf(-acc));
}

// ---------------- kernel 3/6: causal conv(K=4) + SiLU (+ per-head L2 norm for q,k) ----------------

__global__ __launch_bounds__(256) void conv_silu_norm_kernel(
    const float* __restrict__ qpre, const float* __restrict__ kpre, const float* __restrict__ vpre,
    const float* __restrict__ cq, const float* __restrict__ ck, const float* __restrict__ cv,
    float* __restrict__ qo, float* __restrict__ ko, float* __restrict__ vo) {
  const int m   = blockIdx.x;            // row in [B*T]
  const int t   = m & (Tc - 1);          // position within sequence
  const int tid = threadIdx.x;
  const int c0  = tid * 4;               // 4 channels per thread; 16 threads per head

  const float* pres[3] = {qpre, kpre, vpre};
  const float* cws[3]  = {cq, ck, cv};
  float*       outs[3] = {qo, ko, vo};

  #pragma unroll
  for (int s = 0; s < 3; s++) {
    float4 w4[4];
    #pragma unroll
    for (int j = 0; j < 4; j++) w4[j] = *(const float4*)(cws[s] + (size_t)(c0 + j) * 4);

    float y[4] = {0.f, 0.f, 0.f, 0.f};
    #pragma unroll
    for (int i = 0; i < 4; i++) {
      int tt = t - 3 + i;
      if (tt >= 0) {
        float4 xr = *(const float4*)(pres[s] + (size_t)(m - 3 + i) * Dc + c0);
        const float* xp = (const float*)&xr;
        #pragma unroll
        for (int j = 0; j < 4; j++) y[j] = fmaf(xp[j], ((const float*)&w4[j])[i], y[j]);
      }
    }
    #pragma unroll
    for (int j = 0; j < 4; j++) y[j] = silu(y[j]);

    if (s < 2) {  // L2 norm over head dim (64 channels = 16 lanes)
      float ss = y[0]*y[0] + y[1]*y[1] + y[2]*y[2] + y[3]*y[3];
      #pragma unroll
      for (int d = 1; d < 16; d <<= 1) ss += __shfl_xor(ss, d, 32);
      float sc = rsqrtf(ss + 1e-6f);
      #pragma unroll
      for (int j = 0; j < 4; j++) y[j] *= sc;
    }
    *(float4*)(outs[s] + (size_t)m * Dc + c0) = make_float4(y[0], y[1], y[2], y[3]);
  }
}

// ---------------- kernel 4/6: chunked delta-rule scan (chunk=64), fp32 WMMA ----------------

__global__ __launch_bounds__(256) void delta_scan_kernel(
    const float* __restrict__ q, const float* __restrict__ k, const float* __restrict__ v,
    const float* __restrict__ beta, float* __restrict__ o) {
  const int bh = blockIdx.x;              // 32 blocks: (b,h)
  const int b  = bh >> 4, h = bh & 15;
  const int tid = threadIdx.x, lane = tid & 31, wave = tid >> 5;

  __shared__ __align__(16) float S [64][64];  // state dk x dv
  __shared__ __align__(16) float Kc[64][64];  // K chunk (t x dk)
  __shared__ __align__(16) float Qc[64][64];  // Q chunk
  __shared__ __align__(16) float Mm[64][64];  // strict-lower beta*K K^T
  __shared__ __align__(16) float U [64][64];  // rhs -> solved u_t
  __shared__ __align__(16) float At[64][64];  // masked Q K^T (incl diag)
  __shared__ float betas[64];
  __shared__ float red[4][64];

  for (int i = tid; i < 64 * 64; i += 256) (&S[0][0])[i] = 0.f;

  const size_t rowbase = (size_t)b * Tc;
  const int coff = h * DHc;
  const int nL = lane & 15, mOff = (lane >> 4) * 8;

  for (int ch = 0; ch < Tc / 64; ch++) {
    const int tb = ch * 64;
    __syncthreads();   // previous chunk's S-update / Kc readers done before reload

    { // load K,Q chunk + beta (b128 LDS stores)
      int r = tid >> 2, d0 = (tid & 3) * 16;
      const float* qs = q + (rowbase + tb + r) * Dc + coff + d0;
      const float* ks = k + (rowbase + tb + r) * Dc + coff + d0;
      #pragma unroll
      for (int i = 0; i < 4; i++) {
        float4 fq = *(const float4*)(qs + i * 4);
        float4 fk = *(const float4*)(ks + i * 4);
        *(float4*)&Qc[r][d0 + i * 4] = fq;
        *(float4*)&Kc[r][d0 + i * 4] = fk;
      }
      if (tid < 64) betas[tid] = beta[(rowbase + tb + tid) * Hc + h];
    }
    __syncthreads();

    // Mm = strict_lower(beta_t * (K K^T));  At = lower_incl(Q K^T);  U = beta*(V - K S0)
    #pragma unroll
    for (int it = 0; it < 2; it++) {
      int tl = wave * 2 + it, ti = tl >> 2, tj = tl & 3;

      floatx8 am = {};
      floatx8 aa = {};
      floatx8 au = {};
      #pragma unroll
      for (int kk = 0; kk < 64; kk += 4) {
        floatx2 bK = fragB_rows(&Kc[0][0], 64, tj * 16, kk, lane);   // K^T as B
        am = wmma_f32k4(fragA_rows(&Kc[0][0], 64, ti * 16, kk, lane), bK, am);
        aa = wmma_f32k4(fragA_rows(&Qc[0][0], 64, ti * 16, kk, lane), bK, aa);
        au = wmma_f32k4(fragA_rows(&Kc[0][0], 64, ti * 16, kk, lane),
                        fragB_cols(&S[0][0], 64, kk, tj * 16, lane), au);
      }
      #pragma unroll
      for (int r = 0; r < 8; r++) {
        int mm = ti * 16 + mOff + r, nn = tj * 16 + nL;
        Mm[mm][nn] = (nn < mm) ? am[r] * betas[mm] : 0.f;
        At[mm][nn] = (nn <= mm) ? aa[r] : 0.f;
        float vv = v[(rowbase + tb + mm) * Dc + coff + nn];
        U[mm][nn] = betas[mm] * (vv - au[r]);
      }
    }
    __syncthreads();

    // forward substitution: u_t -= sum_{j<t} Mm[t][j] * u_j   (Mm masked/scaled already)
    {
      int cc = tid & 63, jg = tid >> 6;
      for (int t2 = 1; t2 < 64; t2++) {
        float part = 0.f;
        #pragma unroll
        for (int j = jg; j < 64; j += 4) part = fmaf(Mm[t2][j], U[j][cc], part);
        red[jg][cc] = part;
        __syncthreads();
        if (tid < 64) U[t2][tid] -= red[0][tid] + red[1][tid] + red[2][tid] + red[3][tid];
        __syncthreads();
      }
    }

    // O = Q @ S0 + At @ U  -> global
    #pragma unroll
    for (int it = 0; it < 2; it++) {
      int tl = wave * 2 + it, ti = tl >> 2, tj = tl & 3;
      floatx8 acc = {};
      #pragma unroll
      for (int kk = 0; kk < 64; kk += 4)
        acc = wmma_f32k4(fragA_rows(&Qc[0][0], 64, ti * 16, kk, lane),
                         fragB_cols(&S[0][0], 64, kk, tj * 16, lane), acc);
      #pragma unroll
      for (int kk = 0; kk < 64; kk += 4)
        acc = wmma_f32k4(fragA_rows(&At[0][0], 64, ti * 16, kk, lane),
                         fragB_cols(&U[0][0], 64, kk, tj * 16, lane), acc);
      #pragma unroll
      for (int r = 0; r < 8; r++) {
        int mm = ti * 16 + mOff + r, nn = tj * 16 + nL;
        o[(rowbase + tb + mm) * Dc + coff + nn] = acc[r];
      }
    }
    __syncthreads();   // all S0 reads complete before update

    // S += K^T @ U
    #pragma unroll
    for (int it = 0; it < 2; it++) {
      int tl = wave * 2 + it, ti = tl >> 2, tj = tl & 3;
      floatx8 acc;
      #pragma unroll
      for (int r = 0; r < 8; r++) acc[r] = S[ti * 16 + mOff + r][tj * 16 + nL];
      #pragma unroll
      for (int kk = 0; kk < 64; kk += 4)
        acc = wmma_f32k4(fragA_cols(&Kc[0][0], 64, ti * 16, kk, lane),
                         fragB_cols(&U[0][0], 64, kk, tj * 16, lane), acc);
      #pragma unroll
      for (int r = 0; r < 8; r++) S[ti * 16 + mOff + r][tj * 16 + nL] = acc[r];
    }
  }
}

// ---------------- kernel 5/6: per-head RMSNorm ----------------

__global__ __launch_bounds__(256) void rms_kernel(
    const float* __restrict__ o, const float* __restrict__ g, float* __restrict__ on) {
  int m = blockIdx.x, tid = threadIdx.x, c0 = tid * 4;
  float4 y4 = *(const float4*)(o + (size_t)m * Dc + c0);
  float* y = (float*)&y4;
  float ss = y[0]*y[0] + y[1]*y[1] + y[2]*y[2] + y[3]*y[3];
  #pragma unroll
  for (int d = 1; d < 16; d <<= 1) ss += __shfl_xor(ss, d, 32);
  float sc = rsqrtf(ss * (1.f / 64.f) + 1e-6f);
  #pragma unroll
  for (int j = 0; j < 4; j++) y[j] = y[j] * sc * g[(c0 + j) & 63];
  *(float4*)(on + (size_t)m * Dc + c0) = y4;
}

// ---------------- launch ----------------

extern "C" void kernel_launch(void* const* d_in, const int* in_sizes, int n_in,
                              void* d_out, int out_size, void* d_ws, size_t ws_size,
                              hipStream_t stream) {
  const float* x  = (const float*)d_in[0];
  const float* Wq = (const float*)d_in[1];
  const float* Wk = (const float*)d_in[2];
  const float* Wv = (const float*)d_in[3];
  const float* Wb = (const float*)d_in[4];
  const float* cq = (const float*)d_in[5];
  const float* ck = (const float*)d_in[6];
  const float* cv = (const float*)d_in[7];
  const float* rg = (const float*)d_in[8];
  const float* Wo = (const float*)d_in[9];
  float* out = (float*)d_out;

  const size_t MD = (size_t)MTOT * Dc;
  float* ws   = (float*)d_ws;
  float* qpre = ws;             // also reused as o (scan output)
  float* kpre = qpre + MD;      // also reused as o_norm
  float* vpre = kpre + MD;
  float* qb   = vpre + MD;
  float* kb   = qb + MD;
  float* vb   = kb + MD;
  float* bet  = vb + MD;        // [4096 x 16]

  // 1) q/k/v projections (bf16 WMMA, 3 outputs via grid.z)
  gemm_bf16_x3<<<dim3(Dc / 64, MTOT / 128, 3), 256, 0, stream>>>(
      x, Wq, Wk, Wv, qpre, kpre, vpre);
  // 2) beta
  beta_sigmoid_kernel<<<(MTOT * Hc) / 256, 256, 0, stream>>>(x, Wb, bet);
  // 3) conv + silu + l2norm
  conv_silu_norm_kernel<<<MTOT, 256, 0, stream>>>(qpre, kpre, vpre, cq, ck, cv, qb, kb, vb);
  // 4) chunked delta-rule scan (o -> qpre, now free)
  delta_scan_kernel<<<Bc * Hc, 256, 0, stream>>>(qb, kb, vb, bet, qpre);
  // 5) per-head RMSNorm (o_norm -> kpre, now free)
  rms_kernel<<<MTOT, 256, 0, stream>>>(qpre, rg, kpre);
  // 6) output projection
  gemm_bf16_x3<<<dim3(Dc / 64, MTOT / 128, 1), 256, 0, stream>>>(
      kpre, Wo, Wo, Wo, out, out, out);
}
